// BERTLegalCA_90615220011635
// MI455X (gfx1250) — matrix operand order repaired and checked by reading, not compiled, not run on hardware
//
#include <hip/hip_runtime.h>
#include <hip/hip_bf16.h>
#include <cstdint>

#define DI __device__ __forceinline__

static constexpr int L_ = 103, R_ = 512, B_ = 2, S_ = 512, D_ = 768;
static constexpr float SCALE = 0.03608439182435161f; // 1/sqrt(768)

// ---- workspace layout (float-sized offsets; bf16 arrays carved with 16B alignment) ----
static constexpr size_t OFF_GT   = 0;                          // 768*768  G_T[e][c] = sum_d Wk[c,d]*Wq[e,d]
static constexpr size_t OFF_H    = OFF_GT + (size_t)768*768;   // 768      h[e] = bk . Wq[e,:]
static constexpr size_t OFF_VL   = OFF_H  + 768;               // 768      Wv @ w_law_d
static constexpr size_t OFF_VA   = OFF_VL + 768;
static constexpr size_t OFF_VT   = OFF_VA + 768;
static constexpr size_t OFF_VQ   = OFF_VT + 768;               // Wk @ bq
static constexpr size_t OFF_C4   = OFF_VQ + 768;               // 4 consts
static constexpr size_t OFF_K2H  = OFF_C4 + 12;                // B*S*D bf16 (as 393216 floats), 16B aligned
static constexpr size_t OFF_K2L  = OFF_K2H + (size_t)B_*S_*D_/2;
static constexpr size_t OFF_CM   = OFF_K2L + (size_t)B_*S_*D_/2;  // B*S
static constexpr size_t OFF_UL   = OFF_CM + (size_t)B_*S_;
static constexpr size_t OFF_UA   = OFF_UL + (size_t)B_*S_;
static constexpr size_t OFF_UT   = OFF_UA + (size_t)B_*S_;
static constexpr size_t OFF_ACCL = OFF_UT + (size_t)B_*S_;     // B*L each, contiguous x3
static constexpr size_t OFF_ACCA = OFF_ACCL + (size_t)B_*L_;
static constexpr size_t OFF_ACCT = OFF_ACCA + (size_t)B_*L_;

// ---------------- helpers ----------------
DI float dot768(const float* __restrict__ a, const float* __restrict__ b) {
  float s = 0.f;
  #pragma unroll 4
  for (int i = 0; i < 768; i += 4) {
    float4 x = *(const float4*)(a + i);
    float4 y = *(const float4*)(b + i);
    s += x.x*y.x + x.y*y.y + x.z*y.z + x.w*y.w;
  }
  return s;
}

// ---------------- prep kernels ----------------
__global__ __launch_bounds__(256) void k_G(const float* __restrict__ Wk, const float* __restrict__ Wq,
                                           const float* __restrict__ bk,
                                           float* __restrict__ GT, float* __restrict__ h) {
  __shared__ float q[768];
  const int tid = threadIdx.x, e = blockIdx.x;
  for (int i = tid; i < 768; i += 256) q[i] = Wq[(size_t)e*768 + i];
  __syncthreads();
  for (int c = tid; c < 768; c += 256)
    GT[(size_t)e*768 + c] = dot768(Wk + (size_t)c*768, q);
  if (tid == 0) h[e] = dot768(bk, q);
}

__global__ __launch_bounds__(256) void k_vecs(const float* __restrict__ Wv, const float* __restrict__ bv,
                                              const float* __restrict__ Wk, const float* __restrict__ bq,
                                              const float* __restrict__ bk,
                                              const float* __restrict__ wld, const float* __restrict__ wad,
                                              const float* __restrict__ wtd,
                                              float* __restrict__ vL, float* __restrict__ vA,
                                              float* __restrict__ vT, float* __restrict__ vQ,
                                              float* __restrict__ c4) {
  const int id = blockIdx.x * 256 + threadIdx.x;
  if (id < 4 * 768) {
    const int head = id / 768, c = id - head * 768;
    const float* mat = (head < 3) ? Wv : Wk;
    const float* vec = (head == 0) ? wld : (head == 1) ? wad : (head == 2) ? wtd : bq;
    float s = dot768(mat + (size_t)c*768, vec);
    float* dst = (head == 0) ? vL : (head == 1) ? vA : (head == 2) ? vT : vQ;
    dst[c] = s;
  } else if (id < 4 * 768 + 4) {
    const int k = id - 4 * 768;
    const float* a  = (k < 3) ? bv : bq;
    const float* bb = (k == 0) ? wld : (k == 1) ? wad : (k == 2) ? wtd : bk;
    c4[k] = dot768(a, bb);
  }
}

__global__ __launch_bounds__(256) void k_zero(float* __restrict__ acc) {
  const int i = blockIdx.x * 256 + threadIdx.x;
  if (i < 3 * B_ * L_) acc[i] = 0.f;
}

// Per (b,s): K2 row = G_T @ outrow + h, emitted as packed bf16 hi/lo in the exact
// byte order of the 32x16 bf16 B-fragment (K-contiguous), plus u_* scalars and cm.
__global__ __launch_bounds__(256) void k_bs(const float* __restrict__ outp, const int* __restrict__ amask,
                                            const float* __restrict__ GT, const float* __restrict__ h,
                                            const float* __restrict__ vL, const float* __restrict__ vA,
                                            const float* __restrict__ vT, const float* __restrict__ vQ,
                                            const float* __restrict__ c4,
                                            __bf16* __restrict__ K2h, __bf16* __restrict__ K2l,
                                            float* __restrict__ uL, float* __restrict__ uA,
                                            float* __restrict__ uT, float* __restrict__ cm) {
  __shared__ float  orow[768];
  __shared__ float4 red[256];
  const int tid = threadIdx.x, bs = blockIdx.x;     // bs = b*S + s
  const float* src = outp + (size_t)bs * 768;
  for (int i = tid; i < 768; i += 256) orow[i] = src[i];
  __syncthreads();
  for (int e = tid; e < 768; e += 256) {
    float v = dot768(GT + (size_t)e*768, orow) + h[e];
    __bf16 hi = (__bf16)v;
    K2h[(size_t)bs*768 + e] = hi;
    K2l[(size_t)bs*768 + e] = (__bf16)(v - (float)hi);
  }
  float pl = 0.f, pa = 0.f, pt = 0.f, pq = 0.f;
  for (int c = tid; c < 768; c += 256) {
    float o = orow[c];
    pl += o * vL[c]; pa += o * vA[c]; pt += o * vT[c]; pq += o * vQ[c];
  }
  red[tid] = make_float4(pl, pa, pt, pq);
  __syncthreads();
  for (int s = 128; s > 0; s >>= 1) {
    if (tid < s) {
      float4 a = red[tid], b = red[tid + s];
      red[tid] = make_float4(a.x + b.x, a.y + b.y, a.z + b.z, a.w + b.w);
    }
    __syncthreads();
  }
  if (tid == 0) {
    float4 t = red[0];
    uL[bs] = t.x + c4[0]; uA[bs] = t.y + c4[1]; uT[bs] = t.z + c4[2];
    float mz = (float)amask[bs];
    cm[bs] = (t.w + c4[3]) * SCALE + (1.f - mz) * -10000.f;
  }
}

// ---------------- WMMA attention kernel ----------------
typedef __bf16 v16bf __attribute__((ext_vector_type(16)));
typedef float  v8f   __attribute__((ext_vector_type(8)));

union FragU { v16bf v; __bf16 h[16]; };

// 16 fp32 -> hi/lo bf16 A-fragments (K order h[0..15] matches A 16x32 packing)
DI void make_afrag(const float4* __restrict__ q0, const float4* __restrict__ q1, v16bf& hi, v16bf& lo) {
  float f[16];
  float4 a = q0[0], b = q0[1], c = q1[0], d = q1[1];
  f[0]=a.x; f[1]=a.y; f[2]=a.z; f[3]=a.w;  f[4]=b.x; f[5]=b.y; f[6]=b.z; f[7]=b.w;
  f[8]=c.x; f[9]=c.y; f[10]=c.z; f[11]=c.w; f[12]=d.x; f[13]=d.y; f[14]=d.z; f[15]=d.w;
  FragU H, Lw;
  #pragma unroll
  for (int i = 0; i < 16; ++i) {
    __bf16 t = (__bf16)f[i];
    H.h[i]  = t;
    Lw.h[i] = (__bf16)(f[i] - (float)t);
  }
  hi = H.v; lo = Lw.v;
}

__global__ __launch_bounds__(256) void k_attn(const float* __restrict__ laws,
                                              const __bf16* __restrict__ K2h, const __bf16* __restrict__ K2l,
                                              const float* __restrict__ cm,
                                              const float* __restrict__ uL, const float* __restrict__ uA,
                                              const float* __restrict__ uT,
                                              const float* __restrict__ w_law1, const float* __restrict__ w_accu1,
                                              const float* __restrict__ w_term1,
                                              const float* __restrict__ b_law_d, const float* __restrict__ b_accu_d,
                                              const float* __restrict__ b_term_d,
                                              float* __restrict__ accL, float* __restrict__ accA,
                                              float* __restrict__ accT) {
  __shared__ float sc[32][520];                 // 32 rows x 512 scores, padded stride
  __shared__ float uLs[512], uAs[512], uTs[512], cms[512];
  __shared__ float redL[32], redA[32], redT[32];

  const int tid  = threadIdx.x;
  const int mblk = blockIdx.x;                  // 0..15  (32-row block of R)
  const int l    = blockIdx.y;                  // 0..102
  const int b    = blockIdx.z;                  // 0..1

  for (int i = tid; i < 512; i += 256) {
    uLs[i] = uL[b*S_ + i]; uAs[i] = uA[b*S_ + i]; uTs[i] = uT[b*S_ + i]; cms[i] = cm[b*S_ + i];
  }
  __syncthreads();

  const int w = tid >> 5, lane = tid & 31;
  const int half = lane >> 4, lm = lane & 15;
  const int subm = w & 1;                       // which 16-row sub-tile
  const int nbase = (w >> 1) * 128;             // 128 columns of S per wave
  const int m = mblk*32 + subm*16 + lm;         // global r index for A rows

  // Per-lane byte pointers; kb advances them by a constant, j uses constant IOFFSETs.
  const char* pA = (const char*)(laws + ((size_t)l * R_ + m) * D_ + half * 16);
  const char* pH = (const char*)(K2h + ((size_t)b * S_ + nbase + lm) * D_ + half * 16);
  const char* pL = (const char*)(K2l + ((size_t)b * S_ + nbase + lm) * D_ + half * 16);
  constexpr int JSTRIDE = 16 * D_ * 2;          // 24576 bytes between j-tiles (fits s24 ioffset)

  v8f acc[8];
  #pragma unroll
  for (int j = 0; j < 8; ++j)
    #pragma unroll
    for (int i = 0; i < 8; ++i) acc[j][i] = 0.f;

  for (int kb = 0; kb < D_ / 32; ++kb, pA += 32 * 4, pH += 32 * 2, pL += 32 * 2) {
    // A fragment: 16x32 bf16; lane<16 -> K 0..15, lane>=16 -> K 16..31
    v16bf ah, al;
    make_afrag((const float4*)pA, (const float4*)(pA + 32), ah, al);
    #pragma unroll
    for (int j = 0; j < 8; ++j) {
      // B fragment: 32x16 bf16; lane<16 -> K 0..15 of column n, lane>=16 -> K 16..31
      v16bf bh = *(const v16bf*)(pH + j * JSTRIDE);
      v16bf bl = *(const v16bf*)(pL + j * JSTRIDE);
      acc[j] = __builtin_amdgcn_wmma_f32_16x16x32_bf16(false, ah, false, bh, (short)0, acc[j], false, false);
      acc[j] = __builtin_amdgcn_wmma_f32_16x16x32_bf16(false, ah, false, bl, (short)0, acc[j], false, false);
      acc[j] = __builtin_amdgcn_wmma_f32_16x16x32_bf16(false, al, false, bh, (short)0, acc[j], false, false);
    }
  }

  // scatter scores to LDS: C/D layout: VGPR i, lanes 0-15 -> M=i, lanes 16-31 -> M=i+8, N=lane&15
  #pragma unroll
  for (int j = 0; j < 8; ++j) {
    const int col = nbase + j*16 + lm;
    const float cadd = cms[col];
    #pragma unroll
    for (int i = 0; i < 8; ++i) {
      const int row = subm*16 + half*8 + i;
      sc[row][col] = acc[j][i] * SCALE + cadd;
    }
  }
  __syncthreads();

  // fused softmax + 3 weighted sums; 8 lanes per row, stride-8 columns
  {
    const int row = tid >> 3, q = tid & 7;
    float mx = -1e30f;
    #pragma unroll 8
    for (int i = 0; i < 64; ++i) mx = fmaxf(mx, sc[row][q + 8*i]);
    for (int d = 1; d < 8; d <<= 1) mx = fmaxf(mx, __shfl_xor(mx, d, 8));
    float se = 0.f, s0 = 0.f, s1 = 0.f, s2 = 0.f;
    #pragma unroll 4
    for (int i = 0; i < 64; ++i) {
      const int c = q + 8*i;
      const float e = __expf(sc[row][c] - mx);
      se += e; s0 += e * uLs[c]; s1 += e * uAs[c]; s2 += e * uTs[c];
    }
    for (int d = 1; d < 8; d <<= 1) {
      se += __shfl_xor(se, d, 8);
      s0 += __shfl_xor(s0, d, 8);
      s1 += __shfl_xor(s1, d, 8);
      s2 += __shfl_xor(s2, d, 8);
    }
    if (q == 0) {
      const int gm = mblk*32 + row;
      const float inv = 1.f / se;
      redL[row] = tanhf(s0*inv + b_law_d[0])  * w_law1[gm];
      redA[row] = tanhf(s1*inv + b_accu_d[0]) * w_accu1[gm];
      redT[row] = tanhf(s2*inv + b_term_d[0]) * w_term1[gm];
    }
  }
  __syncthreads();
  if (tid == 0) {
    float aL = 0.f, aA = 0.f, aT = 0.f;
    #pragma unroll
    for (int r = 0; r < 32; ++r) { aL += redL[r]; aA += redA[r]; aT += redT[r]; }
    atomicAdd(&accL[b*L_ + l], aL);
    atomicAdd(&accA[b*L_ + l], aA);
    atomicAdd(&accT[b*L_ + l], aT);
  }
}

// ---------------- epilogue ----------------
__global__ __launch_bounds__(256) void k_heads(const float* __restrict__ accL, const float* __restrict__ accA,
                                               const float* __restrict__ accT,
                                               const float* __restrict__ b_law1, const float* __restrict__ b_accu1,
                                               const float* __restrict__ W_accu2, const float* __restrict__ b_accu2,
                                               const float* __restrict__ b_term1, const float* __restrict__ W_term2,
                                               const float* __restrict__ b_term2, float* __restrict__ out) {
  __shared__ float tA[2*103], tT[2*103];
  const int tid = threadIdx.x;
  for (int i = tid; i < 2*103; i += 256) {
    out[i] = accL[i] + b_law1[0];
    tA[i] = tanhf(accA[i] + b_accu1[0]);
    tT[i] = tanhf(accT[i] + b_term1[0]);
  }
  __syncthreads();
  for (int i = tid; i < 2*119; i += 256) {
    const int bb = i / 119, o = i % 119;
    float s = b_accu2[o];
    for (int l2 = 0; l2 < 103; ++l2) s += tA[bb*103 + l2] * W_accu2[o*103 + l2];
    out[206 + i] = s;
  }
  for (int i = tid; i < 2*11; i += 256) {
    const int bb = i / 11, o = i % 11;
    float s = b_term2[o];
    for (int l2 = 0; l2 < 103; ++l2) s += tT[bb*103 + l2] * W_term2[o*103 + l2];
    out[206 + 238 + i] = s;
  }
}

// ---------------- launch ----------------
extern "C" void kernel_launch(void* const* d_in, const int* in_sizes, int n_in,
                              void* d_out, int out_size, void* d_ws, size_t ws_size,
                              hipStream_t stream) {
  (void)in_sizes; (void)n_in; (void)out_size; (void)ws_size;
  const float* output   = (const float*)d_in[0];
  const int*   amask    = (const int*)  d_in[1];
  const float* laws     = (const float*)d_in[2];
  const float* Wq       = (const float*)d_in[3];
  const float* bq       = (const float*)d_in[4];
  const float* Wk       = (const float*)d_in[5];
  const float* bk       = (const float*)d_in[6];
  const float* Wv       = (const float*)d_in[7];
  const float* bv       = (const float*)d_in[8];
  const float* w_law_d  = (const float*)d_in[9];
  const float* b_law_d  = (const float*)d_in[10];
  const float* w_law1   = (const float*)d_in[11];
  const float* b_law1   = (const float*)d_in[12];
  const float* w_accu_d = (const float*)d_in[13];
  const float* b_accu_d = (const float*)d_in[14];
  const float* w_accu1  = (const float*)d_in[15];
  const float* b_accu1  = (const float*)d_in[16];
  const float* W_accu2  = (const float*)d_in[17];
  const float* b_accu2  = (const float*)d_in[18];
  const float* w_term_d = (const float*)d_in[19];
  const float* b_term_d = (const float*)d_in[20];
  const float* w_term1  = (const float*)d_in[21];
  const float* b_term1  = (const float*)d_in[22];
  const float* W_term2  = (const float*)d_in[23];
  const float* b_term2  = (const float*)d_in[24];

  float* ws = (float*)d_ws;
  float* GT = ws + OFF_GT;  float* h  = ws + OFF_H;
  float* vL = ws + OFF_VL;  float* vA = ws + OFF_VA; float* vT = ws + OFF_VT; float* vQ = ws + OFF_VQ;
  float* c4 = ws + OFF_C4;
  __bf16* K2h = (__bf16*)(void*)(ws + OFF_K2H);
  __bf16* K2l = (__bf16*)(void*)(ws + OFF_K2L);
  float* cm = ws + OFF_CM;
  float* uL = ws + OFF_UL;  float* uA = ws + OFF_UA; float* uT = ws + OFF_UT;
  float* accL = ws + OFF_ACCL; float* accA = ws + OFF_ACCA; float* accT = ws + OFF_ACCT;

  k_G   <<<768, 256, 0, stream>>>(Wk, Wq, bk, GT, h);
  k_vecs<<<13, 256, 0, stream>>>(Wv, bv, Wk, bq, bk, w_law_d, w_accu_d, w_term_d, vL, vA, vT, vQ, c4);
  k_zero<<<3, 256, 0, stream>>>(accL);
  k_bs  <<<B_*S_, 256, 0, stream>>>(output, amask, GT, h, vL, vA, vT, vQ, c4, K2h, K2l, uL, uA, uT, cm);
  k_attn<<<dim3(R_/32, L_, B_), 256, 0, stream>>>(laws, K2h, K2l, cm, uL, uA, uT,
                                                  w_law1, w_accu1, w_term1,
                                                  b_law_d, b_accu_d, b_term_d,
                                                  accL, accA, accT);
  k_heads<<<1, 256, 0, stream>>>(accL, accA, accT, b_law1, b_accu1, W_accu2, b_accu2,
                                 b_term1, W_term2, b_term2, (float*)d_out);
}
